// RepeatExploreMechanism_72129680769641
// MI455X (gfx1250) — matrix operand order repaired and checked by reading, not compiled, not run on hardware
//
#include <hip/hip_runtime.h>
#include <math.h>

#define BB 32
#define SS 2048
#define HH 1024

typedef __attribute__((ext_vector_type(16))) __bf16 v16bf;
typedef __attribute__((ext_vector_type(8)))  __bf16 v8bf;
typedef __attribute__((ext_vector_type(8)))  float  v8f;

// Split fp32 into bf16 hi (truncation) + bf16 lo (rounded residual):
// x ~= (float)hi + (float)lo, ~16 mantissa bits captured.
__device__ __forceinline__ void bsplit(float x, __bf16& h, __bf16& l) {
    unsigned u  = __float_as_uint(x);
    unsigned hb = u & 0xffff0000u;
    h = __builtin_bit_cast(__bf16, (unsigned short)(hb >> 16));
    l = (__bf16)(x - __uint_as_float(hb));
}

// Branchless tanh: 1 - 2/(e^{2x}+1).
__device__ __forceinline__ float fast_tanh(float x) {
    float t = __expf(2.0f * x);
    return 1.0f - __fdividef(2.0f, t + 1.0f);
}

// ---------------------------------------------------------------------------
// Kernel 0: pre-split Ure (f32, [H][H]) into persistent bf16 hi/lo planes.
// ---------------------------------------------------------------------------
__global__ void k_splitU(const float* __restrict__ Ure,
                         __bf16* __restrict__ uhi,
                         __bf16* __restrict__ ulo) {
    int i = (blockIdx.x * 256 + threadIdx.x) * 4;
    float4 v = *(const float4*)(Ure + i);
    bsplit(v.x, uhi[i + 0], ulo[i + 0]);
    bsplit(v.y, uhi[i + 1], ulo[i + 1]);
    bsplit(v.z, uhi[i + 2], ulo[i + 2]);
    bsplit(v.w, uhi[i + 3], ulo[i + 3]);
}

// ---------------------------------------------------------------------------
// Kernel 1: last_proj[b][o] = sum_h last_memory[b][h] * Wre[o][h]
// ---------------------------------------------------------------------------
__global__ void k_lastproj(const float* __restrict__ lm,
                           const float* __restrict__ Wre,
                           float* __restrict__ lp) {
    int b = blockIdx.x >> 2;
    int o = ((blockIdx.x & 3) << 8) + threadIdx.x;
    __shared__ float slm[HH];
    for (int h = threadIdx.x; h < HH; h += 256) slm[h] = lm[b * HH + h];
    __syncthreads();
    const float* wr = Wre + (size_t)o * HH;
    float acc = 0.f;
    for (int h = 0; h < HH; h += 4) {
        float4 wv = *(const float4*)(wr + h);
        acc += wv.x * slm[h] + wv.y * slm[h + 1] + wv.z * slm[h + 2] + wv.w * slm[h + 3];
    }
    lp[b * HH + o] = acc;
}

// ---------------------------------------------------------------------------
// Kernel 2: fused  scores[b][s] = Vre . tanh( X[b,s,:] @ Ure^T + last_proj[b] )
// WG = 32 rows, block 512 = 16 waves; wave w owns col tiles {4w..4w+3}.
// kc is the OUTER rolled loop; per step the A hi/lo fragments are loaded once
// from LDS (8 x ds_load_b128) and reused by 24 WMMAs across 4 tiles x 2 row-
// subtiles x 3 split-products, with persistent acc[4][2] (64 VGPRs).
// ---------------------------------------------------------------------------
__global__ __launch_bounds__(512, 1) void k_scores(
                         const float* __restrict__ X,
                         const __bf16* __restrict__ uhi,
                         const __bf16* __restrict__ ulo,
                         const float* __restrict__ lastproj,
                         const float* __restrict__ Vre,
                         float* __restrict__ scores) {
    extern __shared__ char smem[];
    __bf16* sAhi = (__bf16*)smem;                       // 32*1024 bf16
    __bf16* sAlo = (__bf16*)(smem + 32 * HH * 2);       // 32*1024 bf16
    float*  sbuf = (float*)(smem + 2 * 32 * HH * 2);    // 32 floats

    const int tid     = threadIdx.x;
    const int rowBase = blockIdx.x * 32;
    const int b       = rowBase / SS;

    for (int idx = tid; idx < 32 * HH; idx += 512) {
        int r = idx >> 10, k = idx & (HH - 1);
        float x = X[(size_t)(rowBase + r) * HH + k];
        bsplit(x, sAhi[idx], sAlo[idx]);
    }
    if (tid < 32) sbuf[tid] = 0.f;
    __syncthreads();

    const int w    = tid >> 5;
    const int lane = tid & 31;
    const int lm   = lane & 15;
    const int half = lane >> 4;

    // per-wave col tiles and B row pointers
    int colB[4];
    const __bf16* BH[4];
    const __bf16* BL[4];
#pragma unroll
    for (int j = 0; j < 4; ++j) {
        colB[j] = (w * 4 + j) * 16;
        BH[j] = uhi + (size_t)(colB[j] + lm) * HH;
        BL[j] = ulo + (size_t)(colB[j] + lm) * HH;
    }

    v8f acc[4][2];
#pragma unroll
    for (int j = 0; j < 4; ++j) {
        acc[j][0] = {};
        acc[j][1] = {};
    }

#pragma unroll 1
    for (int kc = 0; kc < HH / 32; ++kc) {
        const int k0 = kc * 32;

        // A 16x32 bf16 fragments (ISA layout), shared across all 4 col tiles:
        // row = lm, elems 0..7 -> K=k0+half*8+e, elems 8..15 -> K=k0+16+half*8+e
        v16bf ah[2], al[2];
#pragma unroll
        for (int rs = 0; rs < 2; ++rs) {
            const int arow = rs * 16 + lm;
            const __bf16* pH = sAhi + arow * HH;
            const __bf16* pL = sAlo + arow * HH;
            v8bf h0 = *(const v8bf*)(pH + k0 + half * 8);
            v8bf h1 = *(const v8bf*)(pH + k0 + 16 + half * 8);
            v8bf l0 = *(const v8bf*)(pL + k0 + half * 8);
            v8bf l1 = *(const v8bf*)(pL + k0 + 16 + half * 8);
            ah[rs] = __builtin_shufflevector(h0, h1, 0,1,2,3,4,5,6,7,8,9,10,11,12,13,14,15);
            al[rs] = __builtin_shufflevector(l0, l1, 0,1,2,3,4,5,6,7,8,9,10,11,12,13,14,15);
        }

        const int bo = k0 + half * 8;
#pragma unroll
        for (int j = 0; j < 4; ++j) {
            // B 32x16 bf16: lane -> N = colB[j]+lm, elems 0..7 -> K = bo+e, elems 8..15 -> K = bo+16+(e-8)
            v8bf bh0 = *(const v8bf*)(BH[j] + bo);
            v8bf bh1 = *(const v8bf*)(BH[j] + bo + 16);
            v8bf bl0 = *(const v8bf*)(BL[j] + bo);
            v8bf bl1 = *(const v8bf*)(BL[j] + bo + 16);
            v16bf bh = __builtin_shufflevector(bh0, bh1, 0,1,2,3,4,5,6,7,8,9,10,11,12,13,14,15);
            v16bf bl = __builtin_shufflevector(bl0, bl1, 0,1,2,3,4,5,6,7,8,9,10,11,12,13,14,15);

#pragma unroll
            for (int rs = 0; rs < 2; ++rs) {
                acc[j][rs] = __builtin_amdgcn_wmma_f32_16x16x32_bf16(
                    false, ah[rs], false, bh, (short)0, acc[j][rs], false, false);
                acc[j][rs] = __builtin_amdgcn_wmma_f32_16x16x32_bf16(
                    false, ah[rs], false, bl, (short)0, acc[j][rs], false, false);
                acc[j][rs] = __builtin_amdgcn_wmma_f32_16x16x32_bf16(
                    false, al[rs], false, bh, (short)0, acc[j][rs], false, false);
            }
        }
    }

    // epilogue: + last_proj, tanh, * Vre, per-row partials
    float partial[2][8];
#pragma unroll
    for (int rs = 0; rs < 2; ++rs)
#pragma unroll
        for (int i = 0; i < 8; ++i) partial[rs][i] = 0.f;

#pragma unroll
    for (int j = 0; j < 4; ++j) {
        const int n = colB[j] + lm;
        const float lpv = lastproj[(size_t)b * HH + n];
        const float vr  = Vre[n];
#pragma unroll
        for (int rs = 0; rs < 2; ++rs)
#pragma unroll
            for (int i = 0; i < 8; ++i)
                partial[rs][i] += fast_tanh(acc[j][rs][i] + lpv) * vr;
    }

#pragma unroll
    for (int rs = 0; rs < 2; ++rs)
#pragma unroll
        for (int i = 0; i < 8; ++i) {
            float v = partial[rs][i];
            for (int m = 1; m < 16; m <<= 1) v += __shfl_xor(v, m, 32);
            if (lm == 0) atomicAdd(&sbuf[rs * 16 + half * 8 + i], v);
        }
    __syncthreads();
    if (tid < 32) scores[(size_t)rowBase + tid] = sbuf[tid];
}

// ---------------------------------------------------------------------------
// Kernel 3a: per-batch softmax -> normalized weights wgt[b][s]. grid 32.
// ---------------------------------------------------------------------------
__global__ void k_softmax_stats(const float* __restrict__ scores,
                                float* __restrict__ wgt) {
    int b = blockIdx.x;
    __shared__ float red[256];
    const float* sc = scores + (size_t)b * SS;

    float ls[8];
    float lmax = -INFINITY;
#pragma unroll
    for (int j = 0; j < 8; ++j) {
        float v = sc[j * 256 + threadIdx.x];
        ls[j] = v;
        lmax = fmaxf(lmax, v);
    }
    red[threadIdx.x] = lmax;
    __syncthreads();
    for (int st = 128; st > 0; st >>= 1) {
        if (threadIdx.x < st) red[threadIdx.x] = fmaxf(red[threadIdx.x], red[threadIdx.x + st]);
        __syncthreads();
    }
    float m = red[0];
    __syncthreads();

    float lsum = 0.f;
#pragma unroll
    for (int j = 0; j < 8; ++j) {
        ls[j] = __expf(ls[j] - m);
        lsum += ls[j];
    }
    red[threadIdx.x] = lsum;
    __syncthreads();
    for (int st = 128; st > 0; st >>= 1) {
        if (threadIdx.x < st) red[threadIdx.x] += red[threadIdx.x + st];
        __syncthreads();
    }
    float inv = 1.0f / red[0];
#pragma unroll
    for (int j = 0; j < 8; ++j)
        wgt[(size_t)b * SS + j * 256 + threadIdx.x] = ls[j] * inv;
}

// ---------------------------------------------------------------------------
// Kernel 3b: partial[b*8+sc][h] = sum_{s in chunk} wgt[b][s] * X[b,s,h]
// ---------------------------------------------------------------------------
__global__ void k_ctx_partial(const float* __restrict__ X,
                              const float* __restrict__ wgt,
                              float* __restrict__ partial) {
    int b  = blockIdx.x >> 3;
    int sc = blockIdx.x & 7;
    const float* wb = wgt + (size_t)b * SS + sc * 256;
    const float4* xb = (const float4*)(X + (size_t)b * SS * HH) + (size_t)sc * 256 * (HH / 4);
    float4 acc = make_float4(0.f, 0.f, 0.f, 0.f);
    for (int s = 0; s < 256; ++s) {
        float w = wb[s];
        float4 v = xb[(size_t)s * (HH / 4) + threadIdx.x];
        acc.x += w * v.x; acc.y += w * v.y; acc.z += w * v.z; acc.w += w * v.w;
    }
    ((float4*)partial)[(size_t)blockIdx.x * (HH / 4) + threadIdx.x] = acc;
}

// ---------------------------------------------------------------------------
// Kernel 3c: ctx[b][h] = sum_{sc<8} partial[b*8+sc][h].  grid 32, block 256.
// ---------------------------------------------------------------------------
__global__ void k_ctx_reduce(const float* __restrict__ partial,
                             float* __restrict__ ctx) {
    int b = blockIdx.x;
    float4 sum = make_float4(0.f, 0.f, 0.f, 0.f);
#pragma unroll
    for (int sc = 0; sc < 8; ++sc) {
        float4 v = ((const float4*)partial)[(size_t)(b * 8 + sc) * (HH / 4) + threadIdx.x];
        sum.x += v.x; sum.y += v.y; sum.z += v.z; sum.w += v.w;
    }
    ((float4*)ctx)[(size_t)b * (HH / 4) + threadIdx.x] = sum;
}

// ---------------------------------------------------------------------------
// Kernel 4: logits = ctx @ Wcre^T, softmax over 2.  grid 32, block 256.
// ---------------------------------------------------------------------------
__global__ void k_logits(const float* __restrict__ ctx,
                         const float* __restrict__ Wcre,
                         float* __restrict__ out) {
    int b = blockIdx.x;
    __shared__ float r0[256], r1[256];
    float p0 = 0.f, p1 = 0.f;
    for (int h = threadIdx.x; h < HH; h += 256) {
        float c = ctx[b * HH + h];
        p0 += c * Wcre[h];
        p1 += c * Wcre[HH + h];
    }
    r0[threadIdx.x] = p0;
    r1[threadIdx.x] = p1;
    __syncthreads();
    for (int st = 128; st > 0; st >>= 1) {
        if (threadIdx.x < st) {
            r0[threadIdx.x] += r0[threadIdx.x + st];
            r1[threadIdx.x] += r1[threadIdx.x + st];
        }
        __syncthreads();
    }
    if (threadIdx.x == 0) {
        float a = r0[0], c = r1[0];
        float m = fmaxf(a, c);
        float e0 = __expf(a - m), e1 = __expf(c - m);
        float s = e0 + e1;
        out[b * 2 + 0] = e0 / s;
        out[b * 2 + 1] = e1 / s;
    }
}

extern "C" void kernel_launch(void* const* d_in, const int* in_sizes, int n_in,
                              void* d_out, int out_size, void* d_ws, size_t ws_size,
                              hipStream_t stream) {
    (void)in_sizes; (void)n_in; (void)out_size; (void)ws_size;

    const float* all_memory  = (const float*)d_in[0];
    const float* last_memory = (const float*)d_in[1];
    const float* Wre         = (const float*)d_in[2];
    const float* Ure         = (const float*)d_in[3];
    const float* Vre         = (const float*)d_in[4];
    const float* Wcre        = (const float*)d_in[5];
    float* out = (float*)d_out;

    // workspace layout (floats first, then 16B-aligned bf16 planes)
    float* lp      = (float*)d_ws;            // 32768 f
    float* scores  = lp + BB * HH;            // 65536 f
    float* wgt     = scores + BB * SS;        // 65536 f
    float* ctx     = wgt + BB * SS;           // 32768 f
    float* partial = ctx + BB * HH;           // 262144 f
    __bf16* uhi    = (__bf16*)(partial + BB * 8 * HH);   // offset 1,835,008 B (16B aligned)
    __bf16* ulo    = uhi + (size_t)HH * HH;

    k_splitU<<<(HH * HH) / (256 * 4), 256, 0, stream>>>(Ure, uhi, ulo);
    k_lastproj<<<128, 256, 0, stream>>>(last_memory, Wre, lp);

    size_t dynsmem = (size_t)2 * 32 * HH * 2 + 32 * sizeof(float);  // 128 KB + 128 B
    hipFuncSetAttribute(reinterpret_cast<const void*>(k_scores),
                        hipFuncAttributeMaxDynamicSharedMemorySize, (int)dynsmem);
    k_scores<<<(BB * SS) / 32, 512, dynsmem, stream>>>(all_memory, uhi, ulo, lp, Vre, scores);

    k_softmax_stats<<<BB, 256, 0, stream>>>(scores, wgt);
    k_ctx_partial<<<BB * 8, 256, 0, stream>>>(all_memory, wgt, partial);
    k_ctx_reduce<<<BB, 256, 0, stream>>>(partial, ctx);
    k_logits<<<BB, 256, 0, stream>>>(ctx, Wcre, out);
}
